// SwitchRouter_68272800137426
// MI455X (gfx1250) — matrix operand order repaired and checked by reading, not compiled, not run on hardware
//
#include <hip/hip_runtime.h>
#include <hip/hip_bf16.h>

typedef __attribute__((ext_vector_type(16))) __bf16 v16bf;
typedef __attribute__((ext_vector_type(8)))  __bf16 v8bf;
typedef __attribute__((ext_vector_type(4)))  __bf16 v4bf;
typedef __attribute__((ext_vector_type(8)))  float  v8f;

#define BS_TOK 16384   // B*S
#define DIM    4096
#define NEXP   128
#define MROWS  32               // token rows per workgroup (2 M-tiles)
#define KCHUNK 256
#define NCHUNK (DIM / KCHUNK)   // 16
#define KSTEPS (KCHUNK / 32)    // 8
#define KTOT   (DIM / 32)       // 128

// ---------------------------------------------------------------------------
// Prep: pack gate_weight (D x E, fp32, row-major) into bf16 WMMA B-fragments.
// B (32x16) k order packed here for V_WMMA_F32_16X16X32_BF16 (A uses the same order):
//   lane 0-15  : column N = lane,    elements e=0..15 hold K = e
//   lane 16-31 : column N = lane-16, elements e=0..15 hold K = 16+e
// Stored so each lane reads its 16 bf16 (32B) contiguously.
// Also zeroes the global accumulators (counts[128], psum[128], zsum[1]).
// ---------------------------------------------------------------------------
__global__ void router_pack(const float* __restrict__ gw,
                            __bf16* __restrict__ bp,
                            float* __restrict__ acc) {
  if (blockIdx.x == 0) {
    acc[threadIdx.x] = 0.0f;                // counts[0..127], psum[0..127]
    if (threadIdx.x == 0) acc[256] = 0.0f;  // zsum
  }
  int tid  = blockIdx.x * 256 + threadIdx.x;  // one thread per (nt, ks, lane)
  int nt   = tid >> 12;           // / (128*32)
  int rem  = tid & 4095;
  int ks   = rem >> 5;
  int lane = rem & 31;
  int col   = nt * 16 + (lane & 15);
  int kbase = ks * 32 + ((lane & 16) ? 16 : 0);
  __bf16* dst = bp + ((size_t)((nt * KTOT + ks) * 32 + lane)) * 16;
#pragma unroll
  for (int e = 0; e < 16; ++e)
    dst[e] = (__bf16)gw[(size_t)(kbase + e) * NEXP + col];
}

// ---------------------------------------------------------------------------
// Main fused router: GEMM (bf16 WMMA, fp32 acc) + bias/temp + softmax +
// argmax + prob/weight/expert writes + loss partial accumulation.
// Grid: BS_TOK/32 workgroups, 256 threads (8 waves). Wave w owns expert
// columns [16w, 16w+16); the workgroup owns 32 token rows (2 M-tiles per
// wave, sharing one B fragment per k-step -> 2 WMMAs per B load).
// ---------------------------------------------------------------------------
__global__ __launch_bounds__(256)
void router_main(const float* __restrict__ hs, const __bf16* __restrict__ bp,
                 const float* __restrict__ bias, const float* __restrict__ temp,
                 float* __restrict__ out_rw, float* __restrict__ out_sel,
                 float* __restrict__ out_probs,
                 float* __restrict__ g_counts, float* __restrict__ g_psum,
                 float* __restrict__ g_zsum) {
  __shared__ __attribute__((aligned(16))) char smem[MROWS * KCHUNK * 2]; // 16KB
  __shared__ float Pacc[NEXP];
  __shared__ float zacc;
  __bf16* ldsA   = (__bf16*)smem;        // staged A chunk (32 x KCHUNK bf16)
  float*  logits = (float*)smem;         // reused after GEMM (32 x 128 f32)

  const int t    = threadIdx.x;
  const int lane = t & 31;
  const int wv   = t >> 5;
  const int row0 = blockIdx.x * MROWS;

  if (t < NEXP) Pacc[t] = 0.0f;
  if (t == 0)   zacc = 0.0f;

  v8f acc0 = {};   // rows [0,16) of the block
  v8f acc1 = {};   // rows [16,32)
  const int off0 = (lane & 16) ? 16 : 0;  // A-fragment K-half select per lane

  for (int c = 0; c < NCHUNK; ++c) {
    // ---- stage 32 rows x KCHUNK fp32 of hidden into LDS as bf16 ----
#pragma unroll
    for (int i = 0; i < 8; ++i) {
      int q  = t + i * 256;              // float4 index 0..2047
      int r  = q >> 6;                   // 64 float4 per row
      int kq = (q & 63) << 2;
      const float4 f =
          *(const float4*)&hs[(size_t)(row0 + r) * DIM + c * KCHUNK + kq];
      if (c + 1 < NCHUNK)
        __builtin_prefetch(&hs[(size_t)(row0 + r) * DIM + (c + 1) * KCHUNK + kq], 0, 1);
      v4bf b4 = { (__bf16)f.x, (__bf16)f.y, (__bf16)f.z, (__bf16)f.w };
      *(v4bf*)&ldsA[r * KCHUNK + kq] = b4;
    }
    __syncthreads();
    // ---- 8 k-steps over the chunk, 2 WMMAs per shared B fragment ----
#pragma unroll
    for (int ks = 0; ks < KSTEPS; ++ks) {
      const __bf16* arow0 = &ldsA[(lane & 15) * KCHUNK + ks * 32];
      const __bf16* arow1 = arow0 + 16 * KCHUNK;
      v8bf lo0 = *(const v8bf*)(arow0 + off0);       // ds_load_b128
      v8bf hi0 = *(const v8bf*)(arow0 + off0 + 8);  // ds_load_b128
      v8bf lo1 = *(const v8bf*)(arow1 + off0);
      v8bf hi1 = *(const v8bf*)(arow1 + off0 + 8);
      v16bf af0 = __builtin_shufflevector(lo0, hi0,
          0, 1, 2, 3, 4, 5, 6, 7, 8, 9, 10, 11, 12, 13, 14, 15);
      v16bf af1 = __builtin_shufflevector(lo1, hi1,
          0, 1, 2, 3, 4, 5, 6, 7, 8, 9, 10, 11, 12, 13, 14, 15);
      const v16bf bfrag = *(const v16bf*)(
          bp + ((size_t)((wv * KTOT + (c * KSTEPS + ks)) * 32 + lane)) * 16);
      acc0 = __builtin_amdgcn_wmma_f32_16x16x32_bf16(
          false, af0, false, bfrag, (short)0, acc0, false, false);
      acc1 = __builtin_amdgcn_wmma_f32_16x16x32_bf16(
          false, af1, false, bfrag, (short)0, acc1, false, false);
    }
    __syncthreads();
  }

  // ---- bias + temperature, spill logits to LDS (C layout: VGPR r, lane l ->
  //      row = r + 8*(l>=16), col = l&15; acc1 adds +16 rows) ----
  const int colg = wv * 16 + (lane & 15);
  const float bsv = bias[colg];
  float tv = fminf(fmaxf(temp[colg], 0.1f), 10.0f);
  const int rbase = (lane & 16) ? 8 : 0;
#pragma unroll
  for (int r = 0; r < 8; ++r) {
    logits[(rbase + r) * NEXP + colg]        = (acc0[r] + bsv) / tv;
    logits[(16 + rbase + r) * NEXP + colg]   = (acc1[r] + bsv) / tv;
  }
  __syncthreads();

  // ---- softmax / argmax: 8 lanes per row, 16 cols per lane ----
  const int srow = t >> 3;          // 0..31
  const int sub  = t & 7;           // 0..7
  const float* lrow = &logits[srow * NEXP];
  float lv[16];
#pragma unroll
  for (int j = 0; j < 16; ++j) lv[j] = lrow[sub * 16 + j];

  float bmax = lv[0];
  int   bidx = sub * 16;
#pragma unroll
  for (int j = 1; j < 16; ++j)
    if (lv[j] > bmax) { bmax = lv[j]; bidx = sub * 16 + j; }
#pragma unroll
  for (int m = 1; m < 8; m <<= 1) {
    float om = __shfl_xor(bmax, m, 8);
    int   oi = __shfl_xor(bidx, m, 8);
    if (om > bmax || (om == bmax && oi < bidx)) { bmax = om; bidx = oi; }
  }
  float lsum = 0.0f;
#pragma unroll
  for (int j = 0; j < 16; ++j) lsum += __expf(lv[j] - bmax);
#pragma unroll
  for (int m = 1; m < 8; m <<= 1) lsum += __shfl_xor(lsum, m, 8);
  const float rinv = 1.0f / lsum;

  const long rowG = (long)blockIdx.x * MROWS + srow;
#pragma unroll
  for (int j = 0; j < 16; ++j) {
    float p = __expf(lv[j] - bmax) * rinv;
    out_probs[rowG * NEXP + sub * 16 + j] = p;
    atomicAdd(&Pacc[sub * 16 + j], p);
  }
  if (sub == 0) {
    out_rw[rowG]  = rinv;             // max prob = exp(0)/sum
    out_sel[rowG] = (float)bidx;
    atomicAdd(&g_counts[bidx], 1.0f);
    float z = bmax + __logf(lsum);    // logsumexp
    atomicAdd(&zacc, z * z);
  }
  __syncthreads();
  if (t < NEXP) atomicAdd(&g_psum[t], Pacc[t]);
  if (t == 0)   atomicAdd(g_zsum, zacc);
}

// ---------------------------------------------------------------------------
// Finalize: aux = 0.01 * E * sum(f_i * P_i) + 0.001 * mean(z^2)
// ---------------------------------------------------------------------------
__global__ void router_finalize(const float* __restrict__ g_counts,
                                const float* __restrict__ g_psum,
                                const float* __restrict__ g_zsum,
                                float* __restrict__ out_aux) {
  __shared__ float s[NEXP];
  int t = threadIdx.x;
  s[t] = g_counts[t] * g_psum[t];
  __syncthreads();
  for (int off = 64; off > 0; off >>= 1) {
    if (t < off) s[t] += s[t + off];
    __syncthreads();
  }
  if (t == 0) {
    const float bs = (float)BS_TOK;
    float lb = (float)NEXP * s[0] / (bs * bs);
    float zm = g_zsum[0] / bs;
    out_aux[0] = 0.01f * lb + 0.001f * zm;
  }
}

extern "C" void kernel_launch(void* const* d_in, const int* in_sizes, int n_in,
                              void* d_out, int out_size, void* d_ws, size_t ws_size,
                              hipStream_t stream) {
  const float* hs   = (const float*)d_in[0];  // (4,4096,4096) fp32
  const float* gw   = (const float*)d_in[1];  // (4096,128) fp32
  const float* bias = (const float*)d_in[2];  // (128,) fp32
  const float* temp = (const float*)d_in[3];  // (128,) fp32
  float* out = (float*)d_out;
  // d_out: [0,16384) routing_weights | [16384,32768) selected_experts |
  //        [32768, 32768+16384*128) router_probs | [last] aux_loss
  float* out_rw    = out;
  float* out_sel   = out + BS_TOK;
  float* out_probs = out + 2 * BS_TOK;
  float* out_aux   = out + 2 * BS_TOK + (size_t)BS_TOK * NEXP;

  __bf16* bp  = (__bf16*)d_ws;                        // 1 MB packed B
  float*  acc = (float*)((char*)d_ws + (1u << 20));   // counts|psum|zsum
  float* g_counts = acc;
  float* g_psum   = acc + NEXP;
  float* g_zsum   = acc + 2 * NEXP;

  router_pack<<<128, 256, 0, stream>>>(gw, bp, acc);
  router_main<<<BS_TOK / MROWS, 256, 0, stream>>>(hs, bp, bias, temp,
                                                  out_rw, out_sel, out_probs,
                                                  g_counts, g_psum, g_zsum);
  router_finalize<<<1, NEXP, 0, stream>>>(g_counts, g_psum, g_zsum, out_aux);
}
